// Model_39676907888059
// MI455X (gfx1250) — compile-verified, matrix operand order checked by reading, not run on hardware
//
#include <hip/hip_runtime.h>
#include <hip/hip_bf16.h>
#include <stdint.h>

// ---------------------------------------------------------------------------
// MI455X / gfx1250 fused attention block.
//   qkv = x1 @ W + b ; q *= D^-0.5 ; scores = qk^T + mask ; softmax ; o = p@v
// All matmuls via v_wmma_f32_16x16x32_bf16 (wave32), f32 accumulate.
// 16x4096 f32 score block lives in 256KB of the 320KB WGP LDS.
// Register-blocked: 64x64 per wave in the QKV GEMM (16 WMMA / K-step),
// 16x64 per wave in the QK^T phase (4 WMMA per q-fragment).
// ---------------------------------------------------------------------------

typedef __attribute__((ext_vector_type(16))) __bf16 v16bf;
typedef __attribute__((ext_vector_type(8)))  float  v8f;

#define D_MODEL 1024
#define SEQ     4096
#define BATCH   4
#define N3      3072   // 3*D_MODEL

__device__ __forceinline__ unsigned short f2bf_raw(float f) {
  union { float f; uint32_t u; } v; v.f = f;
  uint32_t r = v.u + 0x7FFFu + ((v.u >> 16) & 1u);   // round-to-nearest-even
  return (unsigned short)(r >> 16);
}
__device__ __forceinline__ __bf16 f2bf(float f) {
  unsigned short h = f2bf_raw(f);
  __bf16 b; __builtin_memcpy(&b, &h, 2); return b;
}

// Load a 16-element bf16 fragment row: elements 0..7 at p, 8..15 at p+16
// (ISA 16-bit A/B layout: lane<16 -> K {0..7,16..23}, lane>=16 -> +8).
__device__ __forceinline__ v16bf load_frag16(const unsigned short* p) {
  uint4 lo = *(const uint4*)p;
  uint4 hi = *(const uint4*)(p + 16);
  v16bf a;
  __builtin_memcpy(&a, &lo, 16);
  __builtin_memcpy((char*)&a + 16, &hi, 16);
  return a;
}

__device__ __forceinline__ v8f wmma_bf16(v16bf a, v16bf b, v8f c) {
  return __builtin_amdgcn_wmma_f32_16x16x32_bf16(false, a, false, b,
                                                 (short)0, c, false, false);
}

// ---------------------------------------------------------------------------
// Kernel 1a: f32 -> bf16 convert (x1)
// ---------------------------------------------------------------------------
__global__ void cvt_bf16_kernel(const float* __restrict__ in,
                                unsigned short* __restrict__ out, int n) {
  int i = blockIdx.x * blockDim.x + threadIdx.x;
  int stride = gridDim.x * blockDim.x;
  for (; i < n; i += stride) out[i] = f2bf_raw(in[i]);
}

// Kernel 1b: W[1024,3072] f32 -> Wt[3072,1024] bf16 (K-contiguous rows)
__global__ void transpose_w_kernel(const float* __restrict__ W,
                                   unsigned short* __restrict__ Wt) {
  int i = blockIdx.x * blockDim.x + threadIdx.x;
  int stride = gridDim.x * blockDim.x;
  const int n = N3 * D_MODEL;
  for (; i < n; i += stride) {
    int nn = i / D_MODEL, kk = i - nn * D_MODEL;
    Wt[i] = f2bf_raw(W[kk * N3 + nn]);
  }
}

// ---------------------------------------------------------------------------
// Kernel 2: QKV GEMM. 64x64 macro-tile per wave: 4 A-frags x 4 B-frags,
// 16 WMMAs per 32-wide K-step (32 steps). Block = 8 waves covers 128x256.
//   q -> qb[BS,1024] (scaled, +bias), k -> kb[BS,1024], v -> vT[B,1024,4096].
// ---------------------------------------------------------------------------
__global__ __launch_bounds__(256, 1) void qkv_gemm_kernel(
    const unsigned short* __restrict__ xb,   // [16384,1024] bf16
    const unsigned short* __restrict__ Wt,   // [3072,1024] bf16
    const float* __restrict__ bqkv,          // [3072] f32
    unsigned short* __restrict__ qb,
    unsigned short* __restrict__ kb,
    unsigned short* __restrict__ vT) {
  const int lane  = threadIdx.x & 31;
  const int wave  = threadIdx.x >> 5;
  const int wm    = wave >> 2;                       // 0..1
  const int wn    = wave & 3;                        // 0..3
  const int m0    = blockIdx.y * 128 + wm * 64;
  const int n0    = blockIdx.x * 256 + wn * 64;
  const int laneM = lane & 15;
  const int off   = (lane < 16) ? 0 : 8;

  const unsigned short* arow[4];
  const unsigned short* brow[4];
#pragma unroll
  for (int i = 0; i < 4; ++i) {
    arow[i] = xb + (size_t)(m0 + i * 16 + laneM) * D_MODEL + off;
    brow[i] = Wt + (size_t)(n0 + i * 16 + laneM) * D_MODEL + off;
  }

  v8f acc[4][4] = {};
  for (int kk = 0; kk < D_MODEL; kk += 32) {
    v16bf a[4], b[4];
#pragma unroll
    for (int i = 0; i < 4; ++i) {
      a[i] = load_frag16(arow[i] + kk);
      b[i] = load_frag16(brow[i] + kk);
    }
#pragma unroll
    for (int mi = 0; mi < 4; ++mi)
#pragma unroll
      for (int ni = 0; ni < 4; ++ni)
        acc[mi][ni] = wmma_bf16(a[mi], b[ni], acc[mi][ni]);
  }

  const int region = n0 >> 10;  // 0=q, 1=k, 2=v (64-tiles never straddle 1024)
#pragma unroll
  for (int ni = 0; ni < 4; ++ni) {
    const int ng   = n0 + ni * 16 + laneM;
    const float bias = bqkv[ng];
    const int nloc = ng & 1023;
#pragma unroll
    for (int mi = 0; mi < 4; ++mi) {
#pragma unroll
      for (int r = 0; r < 8; ++r) {
        int m = m0 + mi * 16 + ((lane < 16) ? r : r + 8);
        float val = acc[mi][ni][r] + bias;
        if (region == 0) {
          qb[(size_t)m * D_MODEL + nloc] = f2bf_raw(val * 0.03125f);
        } else if (region == 1) {
          kb[(size_t)m * D_MODEL + nloc] = f2bf_raw(val);
        } else {
          int bidx = m >> 12, s = m & (SEQ - 1);
          vT[((size_t)bidx * D_MODEL + nloc) * SEQ + s] = f2bf_raw(val);
        }
      }
    }
  }
}

// ---------------------------------------------------------------------------
// Kernel 3: fused attention for one (batch, 16-query) tile per block.
//   LDS: sc[16][4096] f32 scores (256KB) + qs[16][1024] bf16 q tile (32KB).
// ---------------------------------------------------------------------------
__global__ __launch_bounds__(256, 1) void attn_kernel(
    const unsigned short* __restrict__ qb,
    const unsigned short* __restrict__ kb,
    const unsigned short* __restrict__ vT,
    const float* __restrict__ mask,          // [B,S,S] f32, streamed once (NT)
    float* __restrict__ out) {               // [B,S,D] f32
  extern __shared__ char smem[];
  float*          sc = (float*)smem;                                       // 16x4096
  unsigned short* qs = (unsigned short*)(smem + 16 * SEQ * sizeof(float)); // 16x1024

  const int lane  = threadIdx.x & 31;
  const int wave  = threadIdx.x >> 5;
  const int q0    = blockIdx.x * 16;
  const int b     = blockIdx.y;
  const int laneM = lane & 15;
  const int off   = (lane < 16) ? 0 : 8;

  // Phase 0: stage 16x1024 bf16 q tile into LDS (contiguous 32KB copy).
  {
    const uint4* src = (const uint4*)(qb + (size_t)(b * SEQ + q0) * D_MODEL);
    uint4* dst = (uint4*)qs;
    for (int i = threadIdx.x; i < 16 * D_MODEL / 8; i += 256) dst[i] = src[i];
  }
  __syncthreads();

  // Phase 1: scores = q @ k^T + mask. 64 keys (4 tiles) per wave-pass:
  // one LDS A-fragment feeds 4 WMMAs on independent accumulators.
  const unsigned short* arow = qs + (size_t)laneM * D_MODEL + off;
  for (int kt = wave; kt < SEQ / 64; kt += 8) {
    const int k0 = kt * 64;
    const unsigned short* brow[4];
#pragma unroll
    for (int i = 0; i < 4; ++i)
      brow[i] = kb + (size_t)(b * SEQ + k0 + i * 16 + laneM) * D_MODEL + off;
    v8f c[4] = {};
    for (int kk = 0; kk < D_MODEL; kk += 32) {
      v16bf a = load_frag16(arow + kk);
#pragma unroll
      for (int i = 0; i < 4; ++i)
        c[i] = wmma_bf16(a, load_frag16(brow[i] + kk), c[i]);
    }
#pragma unroll
    for (int i = 0; i < 4; ++i) {
#pragma unroll
      for (int r = 0; r < 8; ++r) {
        int m = (lane < 16) ? r : r + 8;
        int n = k0 + i * 16 + laneM;
        float mk = __builtin_nontemporal_load(
            &mask[((size_t)(b * SEQ) + q0 + m) * SEQ + n]);
        sc[m * SEQ + n] = c[i][r] + mk;
      }
    }
  }
  __syncthreads();

  // Phase 2: row softmax in LDS; 2 rows per wave, wave32 shuffle reductions.
  for (int rr = 0; rr < 2; ++rr) {
    int row = wave * 2 + rr;
    float* p = sc + (size_t)row * SEQ;
    float mx = -3.4e38f;
    for (int i = lane; i < SEQ; i += 32) mx = fmaxf(mx, p[i]);
#pragma unroll
    for (int s = 16; s >= 1; s >>= 1) mx = fmaxf(mx, __shfl_xor(mx, s, 32));
    float sum = 0.f;
    for (int i = lane; i < SEQ; i += 32) { float e = __expf(p[i] - mx); p[i] = e; sum += e; }
#pragma unroll
    for (int s = 16; s >= 1; s >>= 1) sum += __shfl_xor(sum, s, 32);
    float inv = 1.f / sum;
    for (int i = lane; i < SEQ; i += 32) p[i] *= inv;
  }
  __syncthreads();

  // Phase 3: o = P @ V. Each wave owns 128 output columns (8 tiles of 16);
  // one rebuilt A-fragment feeds 8 WMMAs.
  const int d0 = wave * 128;
  v8f acc[8] = {};
  for (int ks = 0; ks < SEQ; ks += 32) {
    const float* pp = sc + (size_t)laneM * SEQ + ks + off;
    v16bf a;
#pragma unroll
    for (int j = 0; j < 8; ++j) { a[j] = f2bf(pp[j]); a[8 + j] = f2bf(pp[16 + j]); }
#pragma unroll
    for (int t = 0; t < 8; ++t) {
      const unsigned short* brow =
          vT + ((size_t)(b * D_MODEL) + d0 + t * 16 + laneM) * SEQ + ks + off;
      acc[t] = wmma_bf16(a, load_frag16(brow), acc[t]);
    }
  }
#pragma unroll
  for (int t = 0; t < 8; ++t) {
#pragma unroll
    for (int r = 0; r < 8; ++r) {
      int m = (lane < 16) ? r : r + 8;
      int n = d0 + t * 16 + laneM;
      __builtin_nontemporal_store(
          acc[t][r], &out[((size_t)(b * SEQ) + q0 + m) * D_MODEL + n]);
    }
  }
}

// ---------------------------------------------------------------------------
// Launch. ws layout (bytes):
//   xb  @ 0          : 33,554,432  (16384x1024 bf16)
//   Wt  @ 33554432   :  6,291,456  (3072x1024 bf16)
//   qb  @ 39845888   : 33,554,432
//   kb  @ 73400320   : 33,554,432
//   vT  @ 106954752  : 33,554,432   (total ~140.5 MB)
// ---------------------------------------------------------------------------
extern "C" void kernel_launch(void* const* d_in, const int* in_sizes, int n_in,
                              void* d_out, int out_size, void* d_ws, size_t ws_size,
                              hipStream_t stream) {
  const float* x1   = (const float*)d_in[0];
  const float* mask = (const float*)d_in[2];
  const float* W    = (const float*)d_in[3];
  const float* bq   = (const float*)d_in[4];
  float* out = (float*)d_out;

  char* ws = (char*)d_ws;
  unsigned short* xb = (unsigned short*)(ws);
  unsigned short* Wt = (unsigned short*)(ws + 33554432);
  unsigned short* qb = (unsigned short*)(ws + 39845888);
  unsigned short* kb = (unsigned short*)(ws + 73400320);
  unsigned short* vT = (unsigned short*)(ws + 106954752);

  cvt_bf16_kernel<<<8192, 256, 0, stream>>>(x1, xb, BATCH * SEQ * D_MODEL);
  transpose_w_kernel<<<4096, 256, 0, stream>>>(W, Wt);

  qkv_gemm_kernel<<<dim3(N3 / 256, (BATCH * SEQ) / 128), 256, 0, stream>>>(
      xb, Wt, bq, qb, kb, vT);

  const int smem_bytes = 16 * SEQ * (int)sizeof(float) + 16 * D_MODEL * 2;  // 294,912
  hipFuncSetAttribute(reinterpret_cast<const void*>(attn_kernel),
                      hipFuncAttributeMaxDynamicSharedMemorySize, smem_bytes);
  attn_kernel<<<dim3(SEQ / 16, BATCH), 256, smem_bytes, stream>>>(
      qb, kb, vT, mask, out);
}
